// GeneratorNMP_58523224375721
// MI455X (gfx1250) — matrix operand order repaired and checked by reading, not compiled, not run on hardware
//
#include <hip/hip_runtime.h>
#include <hip/hip_bf16.h>
#include <cstdint>

// ---------------------------------------------------------------------------
// GeneratorNMP on MI455X (gfx1250): all GEMMs via v_wmma_f32_16x16x32_bf16,
// aggregate GEMM stages its B-operand chunks through LDS via the Tensor Data
// Mover (tensor_load_to_lds + s_wait_tensorcnt), double buffered.
// BS=16, N=1024, FEATS=64, HID=128, ITERS=4.  Output: final A [16,1024,1024] f32.
// ---------------------------------------------------------------------------

static constexpr int BS_C    = 16;
static constexpr int NVERT   = 1024;
static constexpr int FEATS_C = 64;
static constexpr int HID_C   = 128;
static constexpr int ITERS_C = 4;
static constexpr int ROWS    = BS_C * NVERT;       // 16384 flat rows

typedef __attribute__((ext_vector_type(16))) __bf16 v16bf;
typedef __attribute__((ext_vector_type(8)))  __bf16 v8bf;
typedef __attribute__((ext_vector_type(8)))  float  v8f;

#if defined(__has_builtin)
#if __has_builtin(__builtin_amdgcn_tensor_load_to_lds) && \
    __has_builtin(__builtin_amdgcn_s_wait_tensorcnt)
#define USE_TDM 1
#endif
#endif
#ifndef USE_TDM
#define USE_TDM 0
#endif

__device__ __forceinline__ v8f wmma_bf16(v16bf a, v16bf b, v8f c) {
  // 8 args: (neg_a, A, neg_b, B, c_mod, C, reuse_a, reuse_b)
  return __builtin_amdgcn_wmma_f32_16x16x32_bf16(false, a, false, b, (short)0, c,
                                                 false, false);
}

// Build the 16x32 bf16 A-operand for one wave.
// Lane L: row M = L&15;  K-slice [off, off+16) with
// off = (L>=16 ? 16 : 0)  (contiguous order, matches B-operand).
__device__ __forceinline__ v16bf make_a_from_bf16(const __bf16* p /*row base + hi*16*/,
                                                  int kc) {
  v8bf lo = *(const v8bf*)(p + kc);
  v8bf hv = *(const v8bf*)(p + kc + 8);
  v16bf a;
#pragma unroll
  for (int i = 0; i < 8; ++i) { a[i] = lo[i]; a[8 + i] = hv[i]; }
  return a;
}

#if USE_TDM
typedef __attribute__((ext_vector_type(4))) unsigned int v4u;
typedef __attribute__((ext_vector_type(8))) int          v8i_t;
typedef __attribute__((ext_vector_type(4))) int          v4i_t;

// TDM: DMA a [tile_dim1=128 rows] x [tile_dim0=32 elems] bf16 tile, row stride
// 1024 elements, from global `gsrc` into LDS at byte offset `lds_addr`.
// Descriptor packing per CDNA5 ISA Sec 8.3/8.4 (D# groups 0 and 1).
__device__ __forceinline__ void tdm_load_msg_tile(const __bf16* gsrc,
                                                  unsigned int lds_addr) {
  uint64_t ga = (uint64_t)(uintptr_t)gsrc;
  v4u g0;
  g0[0] = 1u;                                           // count=1, user D#
  g0[1] = lds_addr;                                     // lds_addr [63:32]
  g0[2] = (unsigned int)ga;                             // global_addr [95:64]
  g0[3] = (unsigned int)((ga >> 32) & 0x1FFFFFFu)       // global_addr [120:96]
        | (2u << 30);                                   // type=2 ("image")
  v8i_t g1;
  g1[0] = (int)(1u << 16);                              // data_size=1 -> 2 bytes
  g1[1] = (int)((1024u & 0xFFFFu) << 16);               // tensor_dim0 lo16
  g1[2] = (int)((1024u >> 16) | (128u << 16));          // tensor_dim0 hi, dim1 lo
  g1[3] = (int)((128u >> 16) | (32u << 16));            // dim1 hi, tile_dim0=32
  g1[4] = (int)128u;                                    // tile_dim1=128, tile_dim2=0
  g1[5] = (int)1024u;                                   // tensor_dim0_stride lo32
  g1[6] = 0;                                            // stride0 hi, stride1 lo
  g1[7] = 0;                                            // stride1 hi
  v4i_t z4 = {0, 0, 0, 0};
#if __clang_major__ >= 23
  v8i_t z8 = {0, 0, 0, 0, 0, 0, 0, 0};
  __builtin_amdgcn_tensor_load_to_lds(g0, g1, z4, z4, z8, 0);
#else
  __builtin_amdgcn_tensor_load_to_lds(g0, g1, z4, z4, 0);
#endif
}
#endif  // USE_TDM

// ---------------------------------------------------------------------------
// f32 -> bf16 elementwise convert
// ---------------------------------------------------------------------------
__global__ void k_f32_to_bf16(const float* __restrict__ in, __bf16* __restrict__ out,
                              int n) {
  int i = blockIdx.x * 256 + threadIdx.x;
  if (i < n) out[i] = (__bf16)in[i];
}

// ---------------------------------------------------------------------------
// Batched transpose: in [z][R][C] (T = float or __bf16) -> out bf16 [z][C][R]
// Classic 32x33 LDS tile (ds_store/ds_load traffic).
// ---------------------------------------------------------------------------
template <typename T>
__global__ void k_transpose_to_bf16(const T* __restrict__ in, __bf16* __restrict__ out,
                                    int R, int C) {
  __shared__ float tile[32][33];
  const T* inb   = in  + (size_t)blockIdx.z * R * C;
  __bf16*  outb  = out + (size_t)blockIdx.z * R * C;
  int c0 = blockIdx.x * 32, r0 = blockIdx.y * 32;
  int tx = threadIdx.x, ty = threadIdx.y;   // block (32, 8)
#pragma unroll
  for (int i = 0; i < 32; i += 8) {
    int r = r0 + ty + i, c = c0 + tx;
    tile[ty + i][tx] = (r < R && c < C) ? (float)inb[(size_t)r * C + c] : 0.0f;
  }
  __syncthreads();
#pragma unroll
  for (int i = 0; i < 32; i += 8) {
    int r = c0 + ty + i, c = r0 + tx;       // out is [C][R]
    if (r < C && c < R) outb[(size_t)r * R + c] = (__bf16)tile[tx][ty + i];
  }
}

// ---------------------------------------------------------------------------
// A0[b][i][j] = |i-j| + 1
// ---------------------------------------------------------------------------
__global__ void k_initA(float* __restrict__ A) {
  size_t t = (size_t)blockIdx.x * 256 + threadIdx.x;
  int j = (int)(t & 1023);
  int i = (int)((t >> 10) & 1023);
  A[t] = fabsf((float)(i - j)) + 1.0f;
}

// ---------------------------------------------------------------------------
// Generic GEMM:  C[bf16, ldc] = relu(A[bf16, lda] @ W + bias),  N = 128 fixed.
// W supplied transposed: WT[n][k], row length K (contiguous B-operand loads).
// One wave -> 16 rows x 128 cols strip; WG = 4 waves = 64 rows.
// ---------------------------------------------------------------------------
__global__ __launch_bounds__(128) void k_gemm_bf16(
    const __bf16* __restrict__ A, int lda,
    const __bf16* __restrict__ WT, int K,
    const float* __restrict__ bias,
    __bf16* __restrict__ C, int ldc) {
  int lane    = threadIdx.x & 31;
  int wave    = threadIdx.x >> 5;
  int rowBase = (blockIdx.x * 4 + wave) * 16;
  int M  = lane & 15;
  int hi = lane >> 4;
  const __bf16* arow = A + (size_t)(rowBase + M) * lda + hi * 16;

  v8f acc[8] = {};
  for (int kc = 0; kc < K; kc += 32) {
    v16bf a  = make_a_from_bf16(arow, kc);
    int   kb = kc + hi * 16;
#pragma unroll
    for (int nt = 0; nt < 8; ++nt) {
      int   n = nt * 16 + M;
      v16bf b = *(const v16bf*)(WT + (size_t)n * K + kb);
      acc[nt] = wmma_bf16(a, b, acc[nt]);
    }
  }
#pragma unroll
  for (int nt = 0; nt < 8; ++nt) {
    int   col = nt * 16 + M;
    float bv  = bias ? bias[col] : 0.0f;
#pragma unroll
    for (int j = 0; j < 8; ++j) {
      int   row = rowBase + j + hi * 8;
      float v   = fmaxf(acc[nt][j] + bv, 0.0f);
      C[(size_t)row * ldc + col] = (__bf16)v;
    }
  }
}

// ---------------------------------------------------------------------------
// Softmax row stats: one wave per row of A (length 1024) -> (rowmax, 1/sumexp)
// ---------------------------------------------------------------------------
__global__ __launch_bounds__(128) void k_softmax_stats(const float* __restrict__ A,
                                                       float2* __restrict__ stats) {
  int gwave = (blockIdx.x * 128 + threadIdx.x) >> 5;   // row id, 0..16383
  int lane  = threadIdx.x & 31;
  const float* row = A + (size_t)gwave * 1024;
  float mx = -3.0e38f;
#pragma unroll 4
  for (int t = 0; t < 32; ++t) mx = fmaxf(mx, row[t * 32 + lane]);
#pragma unroll
  for (int off = 16; off > 0; off >>= 1) mx = fmaxf(mx, __shfl_xor(mx, off, 32));
  float s = 0.0f;
#pragma unroll 4
  for (int t = 0; t < 32; ++t) s += __expf(row[t * 32 + lane] - mx);
#pragma unroll
  for (int off = 16; off > 0; off >>= 1) s += __shfl_xor(s, off, 32);
  if (lane == 0) stats[gwave] = make_float2(mx, 1.0f / s);
}

// ---------------------------------------------------------------------------
// Flash-style aggregate:  m[b,i,:] = softmax(A[b,i,:]) @ msg[b]
// P tile rebuilt per-lane as exp(a - max) * rinv directly into the A-operand.
// The 32x128 msgT chunk shared by all 4 waves is staged into LDS by the
// Tensor Data Mover (double buffered, s_wait_tensorcnt + barrier); B-operands
// then come from ds_load_b128.  Output -> hm[..., 128:256], ldc = 256.
// ---------------------------------------------------------------------------
__global__ __launch_bounds__(128) void k_aggregate(
    const float*  __restrict__ Acur,     // [16,1024,1024]
    const float2* __restrict__ stats,    // [16384]
    const __bf16* __restrict__ msgT,     // [16,128,1024]
    __bf16* __restrict__ hm, int ldc) {
  __shared__ __bf16 sB[2][HID_C * 32];   // 2 x 8 KB double buffer
  int tid     = threadIdx.x;
  int lane    = tid & 31;
  int wave    = tid >> 5;
  int rowBase = (blockIdx.x * 4 + wave) * 16;       // global row
  int b       = rowBase >> 10;
  int M  = lane & 15;
  int hi = lane >> 4;
  int grow = rowBase + M;
  float2 st   = stats[grow];
  float rmax  = st.x, rinv = st.y;
  const float*  arow = Acur + (size_t)grow * 1024 + hi * 16;
  const __bf16* mt   = msgT + (size_t)b * HID_C * NVERT;

  // prologue: stage chunk kc=0 into sB[0]
#if USE_TDM
  if (wave == 0) {
    tdm_load_msg_tile(mt, (unsigned int)(uintptr_t)&sB[0][0]);
    __builtin_amdgcn_s_wait_tensorcnt(0);
  }
#else
  for (int u = tid; u < HID_C * 32 / 8; u += 128) {
    int n = u >> 2, c8 = (u & 3) * 8;
    *(v8bf*)&sB[0][n * 32 + c8] = *(const v8bf*)(mt + (size_t)n * NVERT + c8);
  }
#endif
  __syncthreads();

  v8f acc[8] = {};
  int p = 0;
  for (int kc = 0; kc < NVERT; kc += 32) {
    // prefetch next chunk into the other buffer
    if (kc + 32 < NVERT) {
#if USE_TDM
      if (wave == 0)
        tdm_load_msg_tile(mt + kc + 32, (unsigned int)(uintptr_t)&sB[p ^ 1][0]);
#else
      for (int u = tid; u < HID_C * 32 / 8; u += 128) {
        int n = u >> 2, c8 = (u & 3) * 8;
        *(v8bf*)&sB[p ^ 1][n * 32 + c8] =
            *(const v8bf*)(mt + (size_t)n * NVERT + kc + 32 + c8);
      }
#endif
    }
    // prefetch A stream two chunks ahead (global_prefetch_b8)
    __builtin_prefetch(arow + kc + 64, 0, 1);

    // A-operand: P = exp(a - rowmax) * rinv, straight into bf16 lanes
    float4 f0 = *(const float4*)(arow + kc);
    float4 f1 = *(const float4*)(arow + kc + 4);
    float4 f2 = *(const float4*)(arow + kc + 8);
    float4 f3 = *(const float4*)(arow + kc + 12);
    float t[16] = {f0.x, f0.y, f0.z, f0.w, f1.x, f1.y, f1.z, f1.w,
                   f2.x, f2.y, f2.z, f2.w, f3.x, f3.y, f3.z, f3.w};
    v16bf a;
#pragma unroll
    for (int i = 0; i < 16; ++i) a[i] = (__bf16)(__expf(t[i] - rmax) * rinv);

#pragma unroll
    for (int nt = 0; nt < 8; ++nt) {
      int   n  = nt * 16 + M;
      v16bf bb = *(const v16bf*)&sB[p][n * 32 + hi * 16];
      acc[nt] = wmma_bf16(a, bb, acc[nt]);
    }
#if USE_TDM
    if (wave == 0) __builtin_amdgcn_s_wait_tensorcnt(0);
#endif
    __syncthreads();
    p ^= 1;
  }

#pragma unroll
  for (int nt = 0; nt < 8; ++nt) {
    int col = nt * 16 + M;
#pragma unroll
    for (int j = 0; j < 8; ++j) {
      int row = rowBase + j + hi * 8;
      hm[(size_t)row * ldc + HID_C + col] = (__bf16)acc[nt][j];
    }
  }
}

// ---------------------------------------------------------------------------
// Gram: A_out[b,i,j] = scale * dot(h[b,i,:], h[b,j,:]).  B-operand reads
// h rows contiguously along K (h plays its own transpose).
// grid: (256 row-tile blocks, 8 col strips of 128)
// ---------------------------------------------------------------------------
__global__ __launch_bounds__(128) void k_gram(
    const __bf16* __restrict__ H, int ldh,   // hm_next, use cols [0,128)
    float* __restrict__ Aout, float scale) {
  int lane    = threadIdx.x & 31;
  int wave    = threadIdx.x >> 5;
  int rowBase = (blockIdx.x * 4 + wave) * 16;       // global row
  int b       = rowBase >> 10;
  int M  = lane & 15;
  int hi = lane >> 4;
  const __bf16* arow = H + (size_t)(rowBase + M) * ldh + hi * 16;

  v8f acc[8] = {};
  for (int kc = 0; kc < HID_C; kc += 32) {
    v16bf a  = make_a_from_bf16(arow, kc);
    int   kb = kc + hi * 16;
#pragma unroll
    for (int nt = 0; nt < 8; ++nt) {
      int ncol = blockIdx.y * 128 + nt * 16 + M;          // col within batch
      const __bf16* brow = H + (size_t)(b * NVERT + ncol) * ldh;
      v16bf bb = *(const v16bf*)(brow + kb);
      acc[nt] = wmma_bf16(a, bb, acc[nt]);
    }
  }
#pragma unroll
  for (int nt = 0; nt < 8; ++nt) {
    int col = blockIdx.y * 128 + nt * 16 + M;
#pragma unroll
    for (int j = 0; j < 8; ++j) {
      int row = rowBase + j + hi * 8;
      Aout[(size_t)row * 1024 + col] = acc[nt][j] * scale;
    }
  }
}

// ---------------------------------------------------------------------------
// host-side orchestration
// ---------------------------------------------------------------------------
extern "C" void kernel_launch(void* const* d_in, const int* in_sizes, int n_in,
                              void* d_out, int out_size, void* d_ws, size_t ws_size,
                              hipStream_t stream) {
  (void)in_sizes; (void)n_in; (void)out_size; (void)ws_size;
  const float* x   = (const float*)d_in[0];   // [16,1024,64]
  const float* We0 = (const float*)d_in[1];   // [64,128]
  const float* be0 = (const float*)d_in[2];   // [128]
  const float* We1 = (const float*)d_in[3];   // [128,128]
  const float* be1 = (const float*)d_in[4];   // [128]
  const float* Wm  = (const float*)d_in[5];   // [4,128,128]
  const float* bm  = (const float*)d_in[6];   // [4,128]
  const float* Wu  = (const float*)d_in[7];   // [4,256,128]
  const float* bu  = (const float*)d_in[8];   // [4,128]

  // workspace layout
  char* w = (char*)d_ws;
  auto take = [&](size_t bytes) { char* p = w; w += (bytes + 255) & ~(size_t)255; return p; };
  float*   A_cur = (float*)  take((size_t)BS_C * NVERT * NVERT * 4);       // 64 MB
  __bf16*  hm0   = (__bf16*) take((size_t)ROWS * 2 * HID_C * 2);           // 8 MB
  __bf16*  hm1   = (__bf16*) take((size_t)ROWS * 2 * HID_C * 2);           // 8 MB
  __bf16*  msg   = (__bf16*) take((size_t)ROWS * HID_C * 2);               // 4 MB
  __bf16*  msgT  = (__bf16*) take((size_t)BS_C * HID_C * NVERT * 2);       // 4 MB
  __bf16*  tmp_e = (__bf16*) take((size_t)ROWS * HID_C * 2);               // 4 MB
  __bf16*  x_bf  = (__bf16*) take((size_t)ROWS * FEATS_C * 2);             // 2 MB
  __bf16*  We0T  = (__bf16*) take((size_t)HID_C * FEATS_C * 2);
  __bf16*  We1T  = (__bf16*) take((size_t)HID_C * HID_C * 2);
  __bf16*  WmT   = (__bf16*) take((size_t)ITERS_C * HID_C * HID_C * 2);
  __bf16*  WuT   = (__bf16*) take((size_t)ITERS_C * HID_C * 2 * HID_C * 2);
  float2*  stats = (float2*) take((size_t)ROWS * sizeof(float2));

  dim3 tb(32, 8);

  // input/weight preprocessing: x -> bf16, W -> W^T bf16
  k_f32_to_bf16<<<dim3(ROWS * FEATS_C / 256), dim3(256), 0, stream>>>(x, x_bf, ROWS * FEATS_C);
  k_transpose_to_bf16<float><<<dim3(4, 2, 1),       tb, 0, stream>>>(We0, We0T, FEATS_C, HID_C);
  k_transpose_to_bf16<float><<<dim3(4, 4, 1),       tb, 0, stream>>>(We1, We1T, HID_C, HID_C);
  k_transpose_to_bf16<float><<<dim3(4, 4, ITERS_C), tb, 0, stream>>>(Wm, WmT, HID_C, HID_C);
  k_transpose_to_bf16<float><<<dim3(4, 8, ITERS_C), tb, 0, stream>>>(Wu, WuT, 2 * HID_C, HID_C);

  // embedding MLP
  k_gemm_bf16<<<dim3(ROWS / 64), dim3(128), 0, stream>>>(x_bf, FEATS_C, We0T, FEATS_C, be0, tmp_e, HID_C);
  k_gemm_bf16<<<dim3(ROWS / 64), dim3(128), 0, stream>>>(tmp_e, HID_C, We1T, HID_C, be1, hm0, 2 * HID_C);

  // A0 = |i-j| + 1 (broadcast over batch)
  k_initA<<<dim3((unsigned)((size_t)BS_C * NVERT * NVERT / 256)), dim3(256), 0, stream>>>(A_cur);

  __bf16* hmc = hm0;
  __bf16* hmn = hm1;
  const float inv_sqrt_d = 0.08838834764831845f;   // 1/sqrt(128)

  for (int k = 0; k < ITERS_C; ++k) {
    // msg = relu(h @ Wm_k + bm_k)
    k_gemm_bf16<<<dim3(ROWS / 64), dim3(128), 0, stream>>>(
        hmc, 2 * HID_C, WmT + (size_t)k * HID_C * HID_C, HID_C, bm + k * HID_C, msg, HID_C);
    // msgT per batch
    k_transpose_to_bf16<__bf16><<<dim3(4, 32, BS_C), tb, 0, stream>>>(msg, msgT, NVERT, HID_C);
    // softmax row stats on A
    k_softmax_stats<<<dim3(ROWS / 4), dim3(128), 0, stream>>>(A_cur, stats);
    // m = softmax(A) @ msg  -> hm[..., 128:256]
    k_aggregate<<<dim3(ROWS / 64), dim3(128), 0, stream>>>(A_cur, stats, msgT, hmc, 2 * HID_C);
    // h_new = relu([h, m] @ Wu_k + bu_k)
    k_gemm_bf16<<<dim3(ROWS / 64), dim3(128), 0, stream>>>(
        hmc, 2 * HID_C, WuT + (size_t)k * HID_C * 2 * HID_C, 2 * HID_C, bu + k * HID_C,
        hmn, 2 * HID_C);
    // A_new = h_new h_new^T / sqrt(d)
    float* Adst = (k == ITERS_C - 1) ? (float*)d_out : A_cur;
    k_gram<<<dim3(ROWS / 64, 8), dim3(128), 0, stream>>>(hmn, 2 * HID_C, Adst, inv_sqrt_d);

    __bf16* t = hmc; hmc = hmn; hmn = t;
  }
}
